// ProgramPredictor_17849884082916
// MI455X (gfx1250) — matrix operand order repaired and checked by reading, not compiled, not run on hardware
//
#include <hip/hip_runtime.h>

typedef __bf16 v8bf  __attribute__((ext_vector_type(8)));
typedef __bf16 v16bf __attribute__((ext_vector_type(16)));
typedef float  v8f   __attribute__((ext_vector_type(8)));

#define DIM     1024
#define BSZ     512
#define ODIM    5377
#define NHEADS  10
#define NLIN    5

union BF16Frag { v16bf v; v8bf h[2]; };

// ---------------------------------------------------------------------------
// fp32 -> bf16 elementwise conversion (grid-stride)
// ---------------------------------------------------------------------------
__global__ void cvt_f32_bf16(const float* __restrict__ src,
                             __bf16* __restrict__ dst, int n) {
  int i = blockIdx.x * blockDim.x + threadIdx.x;
  int stride = gridDim.x * blockDim.x;
  for (; i < n; i += stride) dst[i] = (__bf16)src[i];
}

// ---------------------------------------------------------------------------
// Group the 512 batch rows by selected head (single block, 512 threads).
// Order within a head is irrelevant: each output row depends only on itself.
// ---------------------------------------------------------------------------
__global__ void build_lists(const int* __restrict__ rule_len,
                            int* __restrict__ lists,
                            int* __restrict__ counts) {
  __shared__ int cnt[NHEADS];
  int t = threadIdx.x;
  if (t < NHEADS) cnt[t] = 0;
  __syncthreads();
  int l = rule_len[t];
  int pos = atomicAdd(&cnt[l], 1);
  lists[l * BSZ + pos] = t;
  __syncthreads();
  if (t < NHEADS) counts[t] = cnt[t];
}

// ---------------------------------------------------------------------------
// MLP layer GEMM: out[m,n] = sum_k A[m,k] * W[n,k] + bias[n]
// A: 512x1024 bf16, W: 1024x1024 bf16 (row-major [n,k]), out: bf16.
//
// Occupancy-oriented tiling: block = 256 threads (8 waves), block tile
// 32(M) x 128(N); waves arranged 2(M) x 4(N), each wave owns 16x32 (2 WMMA
// n-tiles, only 2 f32 accumulators -> low VGPR pressure). Grid = 8 x 16 =
// 128 blocks / 1024 waves so the tiny (3 MB/layer) GEMM is latency-hidden.
// B (=W rows) staged through double-buffered LDS: one barrier per K-step,
// the coop-load of K+32 is issued before the WMMAs on K.
// ---------------------------------------------------------------------------
__global__ __launch_bounds__(256)
void gemm_mlp(const __bf16* __restrict__ A, const __bf16* __restrict__ W,
              const float* __restrict__ bias, __bf16* __restrict__ out) {
  __shared__ alignas(32) __bf16 shB[2][128 * 32];   // 2 x 8 KB
  const int lane = threadIdx.x & 31;
  const int wave = threadIdx.x >> 5;
  const int wm = wave & 1;                 // 2 wave-rows (M)
  const int wn = wave >> 1;                // 4 wave-cols (N, 32 each)
  const int n0 = blockIdx.x * 128;
  const int m0 = blockIdx.y * 32 + wm * 16;

  v8f acc[2] = {};

  const int arow  = m0 + (lane & 15);
  const int akoff = (lane >> 4) ? 16 : 0;  // A-frag K half per lane group
  const int nl = threadIdx.x >> 1;         // coop-load row   (0..127)
  const int hh = threadIdx.x & 1;          // coop-load K half

  // prologue: stage K=0 tile into buffer 0
  {
    const v16bf* src = reinterpret_cast<const v16bf*>(
        W + (size_t)(n0 + nl) * DIM + hh * 16);
    *reinterpret_cast<v16bf*>(&shB[0][nl * 32 + hh * 16]) = *src;
  }

  int cur = 0;
  for (int k0 = 0; k0 < DIM; k0 += 32) {
    __syncthreads();
    // stage next K tile into the other buffer while computing this one
    if (k0 + 32 < DIM) {
      const v16bf* src = reinterpret_cast<const v16bf*>(
          W + (size_t)(n0 + nl) * DIM + (k0 + 32) + hh * 16);
      *reinterpret_cast<v16bf*>(&shB[cur ^ 1][nl * 32 + hh * 16]) = *src;
    }

    // A fragment (16x32): lane<16 holds K 0..15, lane>=16 K 16..31
    BF16Frag a;
    const __bf16* abase = A + (size_t)arow * DIM + k0 + akoff;
    a.h[0] = *reinterpret_cast<const v8bf*>(abase);
    a.h[1] = *reinterpret_cast<const v8bf*>(abase + 8);

#pragma unroll
    for (int nt = 0; nt < 2; ++nt) {
      // B fragment (32x16): lane<16 -> N=lane, K 0..15 ; lane>=16 -> N=lane-16, K 16..31
      const __bf16* bbase =
          &shB[cur][((wn * 2 + nt) * 16 + (lane & 15)) * 32 +
                    ((lane >> 4) ? 16 : 0)];
      v16bf b = *reinterpret_cast<const v16bf*>(bbase);
      acc[nt] = __builtin_amdgcn_wmma_f32_16x16x32_bf16(
          false, a.v, false, b, (short)0, acc[nt], false, false);
    }
    cur ^= 1;
  }

  // epilogue: bias add, store bf16 activations for the next layer
  const int mrow = m0 + ((lane >> 4) ? 8 : 0);
#pragma unroll
  for (int nt = 0; nt < 2; ++nt) {
    int n = n0 + (wn * 2 + nt) * 16 + (lane & 15);
    float bn = bias[n];
#pragma unroll
    for (int i = 0; i < 8; ++i) {
      int m = mrow + i;
      out[(size_t)m * DIM + n] = (__bf16)(acc[nt][i] + bn);
    }
  }
}

// ---------------------------------------------------------------------------
// Head projection: for each head l, for the batch rows selecting l:
//   out[b, o] = sum_k h[b,k] * W_heads[l,o,k] + b_heads[l,o]
// W_heads stays fp32 in HBM (read exactly once, ~220 MB -> the 9.5us BW floor)
// and is converted to bf16 in-register to feed WMMA; batch rows gathered via
// lists[]. Lane mapping consumes full 128B lines of each weight row: lanes
// 0-15 read bytes [0,64) and lanes 16-31 bytes [64,128) of the same rows.
// Block = 256 threads (8 waves): wave w handles M-tile (blockIdx.y*8+w)*16 of
// the head's row list and a 16x64 output strip (4 WMMA o-tiles).
// ---------------------------------------------------------------------------
__global__ __launch_bounds__(256)
void head_gemm(const __bf16* __restrict__ H, const float* __restrict__ Wh,
               const float* __restrict__ bh, const int* __restrict__ lists,
               const int* __restrict__ counts, float* __restrict__ out) {
  const int lane = threadIdx.x & 31;
  const int wave = threadIdx.x >> 5;
  const int head = blockIdx.z;
  const int cnt  = counts[head];
  const int m0   = (blockIdx.y * 8 + wave) * 16;
  if (m0 >= cnt) return;                      // inactive M-tile for this head
  const int o0 = blockIdx.x * 64;
  const int* rows = lists + head * BSZ;

  // gathered activation row for this lane's A-fragment position
  int amidx = m0 + (lane & 15);
  if (amidx >= cnt) amidx = cnt - 1;          // replicate last valid row
  const int arow  = rows[amidx];
  const int akoff = (lane >> 4) ? 16 : 0;

  const size_t wbase = (size_t)head * ODIM * DIM;
  const int bk_half = (lane >> 4) ? 16 : 0;   // B-frag K half per lane group

  v8f acc[4] = {};

  for (int k0 = 0; k0 < DIM; k0 += 32) {
    BF16Frag a;
    const __bf16* abase = H + (size_t)arow * DIM + k0 + akoff;
    a.h[0] = *reinterpret_cast<const v8bf*>(abase);
    a.h[1] = *reinterpret_cast<const v8bf*>(abase + 8);

#pragma unroll
    for (int ot = 0; ot < 4; ++ot) {
      int o  = o0 + ot * 16 + (lane & 15);
      int oc = (o < ODIM) ? o : (ODIM - 1);
      const float4* p = reinterpret_cast<const float4*>(
          Wh + wbase + (size_t)oc * DIM + k0 + bk_half);
      if (k0 + 32 < DIM)                      // stream-ahead hint for W_heads
        __builtin_prefetch(p + 8, 0, 1);
      float4 f0 = p[0], f1 = p[1], f2 = p[2], f3 = p[3];
      BF16Frag b;
      b.v[0]  = (__bf16)f0.x; b.v[1]  = (__bf16)f0.y;
      b.v[2]  = (__bf16)f0.z; b.v[3]  = (__bf16)f0.w;
      b.v[4]  = (__bf16)f1.x; b.v[5]  = (__bf16)f1.y;
      b.v[6]  = (__bf16)f1.z; b.v[7]  = (__bf16)f1.w;
      b.v[8]  = (__bf16)f2.x; b.v[9]  = (__bf16)f2.y;
      b.v[10] = (__bf16)f2.z; b.v[11] = (__bf16)f2.w;
      b.v[12] = (__bf16)f3.x; b.v[13] = (__bf16)f3.y;
      b.v[14] = (__bf16)f3.z; b.v[15] = (__bf16)f3.w;
      acc[ot] = __builtin_amdgcn_wmma_f32_16x16x32_bf16(
          false, a.v, false, b.v, (short)0, acc[ot], false, false);
    }
  }

  // epilogue: bias + scattered fp32 stores (row stride 5377)
#pragma unroll
  for (int ot = 0; ot < 4; ++ot) {
    int o = o0 + ot * 16 + (lane & 15);
    if (o >= ODIM) continue;
    float bno = bh[(size_t)head * ODIM + o];
#pragma unroll
    for (int i = 0; i < 8; ++i) {
      int mi = m0 + i + ((lane >> 4) ? 8 : 0);
      if (mi < cnt) {
        int b = rows[mi];
        out[(size_t)b * ODIM + o] = acc[ot][i] + bno;
      }
    }
  }
}

// ---------------------------------------------------------------------------
extern "C" void kernel_launch(void* const* d_in, const int* in_sizes, int n_in,
                              void* d_out, int out_size, void* d_ws,
                              size_t ws_size, hipStream_t stream) {
  (void)in_sizes; (void)n_in; (void)out_size; (void)ws_size;
  const float* latent   = (const float*)d_in[0];
  const float* W_mlp    = (const float*)d_in[1];
  const float* b_mlp    = (const float*)d_in[2];
  const float* W_heads  = (const float*)d_in[3];
  const float* b_heads  = (const float*)d_in[4];
  const int*   rule_len = (const int*)d_in[5];
  float* out = (float*)d_out;

  // workspace layout (~12.5 MB)
  char* ws = (char*)d_ws;
  __bf16* Wbf  = (__bf16*)ws;                       // 5*1024*1024 bf16
  size_t off = (size_t)NLIN * DIM * DIM * 2;
  __bf16* actA = (__bf16*)(ws + off); off += (size_t)BSZ * DIM * 2;
  __bf16* actB = (__bf16*)(ws + off); off += (size_t)BSZ * DIM * 2;
  int* lists   = (int*)(ws + off);    off += (size_t)NHEADS * BSZ * 4;
  int* counts  = (int*)(ws + off);    off += 64;

  // phase 0: conversions + head grouping
  cvt_f32_bf16<<<2048, 256, 0, stream>>>(W_mlp, Wbf, NLIN * DIM * DIM);
  cvt_f32_bf16<<<512, 256, 0, stream>>>(latent, actA, BSZ * DIM);
  build_lists<<<1, BSZ, 0, stream>>>(rule_len, lists, counts);

  // phase 1: 5 chained MLP GEMMs (ping-pong activations)
  // grid: 8 (N/128) x 16 (M/32) = 128 blocks, 1024 waves
  __bf16* a = actA;
  __bf16* b = actB;
  for (int l = 0; l < NLIN; ++l) {
    gemm_mlp<<<dim3(DIM / 128, BSZ / 32), 256, 0, stream>>>(
        a, Wbf + (size_t)l * DIM * DIM, b_mlp + l * DIM, b);
    __bf16* t = a; a = b; b = t;
  }

  // phase 2: selected-head projection, grouped by head
  // grid: x = ceil(5377/64)=85 o-strips, y = 4 (x8 waves = 32 M-tiles), z = 10 heads
  head_gemm<<<dim3(85, 4, NHEADS), 256, 0, stream>>>(
      a, W_heads, b_heads, lists, counts, out);
}
